// Discriminator_23235773071434
// MI455X (gfx1250) — compile-verified, matrix operand order checked by reading, not run on hardware
//
#include <hip/hip_runtime.h>
#include <hip/hip_bf16.h>

typedef __attribute__((ext_vector_type(16))) _Float16 v16h;
typedef __attribute__((ext_vector_type(8)))  _Float16 v8h;
typedef __attribute__((ext_vector_type(8)))  float    v8f;

#define N_NODES 50000
#define N_EDGES 800000
#define N_GRAPHS 32

__device__ __forceinline__ float leakyf(float v) { return v >= 0.0f ? v : 0.2f * v; }

// ---------------------------------------------------------------------------
// Zero fill
// ---------------------------------------------------------------------------
__global__ void zero_f32(float* __restrict__ p, long long n) {
    long long i = (long long)blockIdx.x * blockDim.x + threadIdx.x;
    if (i < n) p[i] = 0.0f;
}

// ---------------------------------------------------------------------------
// Degree count: deg[dst[e]] += 1
// ---------------------------------------------------------------------------
__global__ void deg_count(const int* __restrict__ dst, float* __restrict__ deg, int E) {
    int e = blockIdx.x * blockDim.x + threadIdx.x;
    if (e < E) atomicAdd(&deg[dst[e]], 1.0f);
}

// ---------------------------------------------------------------------------
// Pack fp32 weight matrix W[Krows x Nout] (zero-padded to Kpad rows) into the
// per-lane f16 WMMA B-fragment layout:
//   fragment (kt, nt): 32 lanes x 16 halves contiguous.
//   lane: n = nt*16 + (lane&15); kbase = (lane>>4)*8
//   half e<8 : K = kt*32 + kbase + e
//   half e>=8: K = kt*32 + 16 + kbase + (e-8)
// ---------------------------------------------------------------------------
__global__ void pack_w_f16(const float* __restrict__ W, int Krows, int Nout, int Kpad,
                           _Float16* __restrict__ out) {
    int t = blockIdx.x * blockDim.x + threadIdx.x;
    int nNT = Nout >> 4;
    int nKT = Kpad >> 5;
    int total = nKT * nNT * 32;
    if (t >= total) return;
    int lane = t & 31;
    int tile = t >> 5;
    int nt = tile % nNT;
    int kt = tile / nNT;
    int n = nt * 16 + (lane & 15);
    int kbase = (lane >> 4) * 8;
    _Float16* dst = out + (size_t)t * 16;
#pragma unroll
    for (int e = 0; e < 16; ++e) {
        int kk = (e < 8) ? (kbase + e) : (16 + kbase + (e - 8));
        int K = kt * 32 + kk;
        float v = (K < Krows) ? W[(size_t)K * Nout + n] : 0.0f;
        dst[e] = (_Float16)v;
    }
}

// ---------------------------------------------------------------------------
// Activation fp32 [M x srcDim] -> f16 row-major [M x Kpad] (zero pad)
// ---------------------------------------------------------------------------
__global__ void prep_a_f16(const float* __restrict__ src, int srcDim, int Kpad,
                           _Float16* __restrict__ A) {
    int r = blockIdx.x;
    for (int k = threadIdx.x; k < Kpad; k += blockDim.x) {
        float v = (k < srcDim) ? src[(size_t)r * srcDim + k] : 0.0f;
        A[(size_t)r * Kpad + k] = (_Float16)v;
    }
}

// ---------------------------------------------------------------------------
// Update-GEMM input: A = [ agg*inv_deg | h ] -> f16 [M x Kpad] (fused mean)
// ---------------------------------------------------------------------------
__global__ void prep_au_f16(const float* __restrict__ agg, const float* __restrict__ deg,
                            const float* __restrict__ h, int hDim, int aggDim, int Kpad,
                            _Float16* __restrict__ A) {
    int r = blockIdx.x;
    float inv = 1.0f / fmaxf(deg[r], 1.0f);
    for (int k = threadIdx.x; k < Kpad; k += blockDim.x) {
        float v;
        if (k < aggDim)             v = agg[(size_t)r * aggDim + k] * inv;
        else if (k < aggDim + hDim) v = h[(size_t)r * hDim + (k - aggDim)];
        else                        v = 0.0f;
        A[(size_t)r * Kpad + k] = (_Float16)v;
    }
}

// ---------------------------------------------------------------------------
// WMMA GEMM: C[M x Nout] = (leaky?)(A16[M x Kpad] @ Wpacked + bias)
// One wave per 16x64 C tile (4 N-tiles): per K-step, 1 A fragment is shared by
// 4 back-to-back v_wmma_f32_16x16x32_f16 against 4 contiguous B fragments.
// M mult of 16, Nout mult of 64, Kpad mult of 32.
// ---------------------------------------------------------------------------
__global__ __launch_bounds__(256) void gemm_wmma_f16(
    const _Float16* __restrict__ A, const _Float16* __restrict__ Wp,
    const float* __restrict__ bias, float* __restrict__ C,
    int M, int Kpad, int Nout, int leakyFlag)
{
    int lane = threadIdx.x & 31;
    int wave = threadIdx.x >> 5;
    int nNT = Nout >> 4;          // 16-col tiles
    int nNG = Nout >> 6;          // 64-col groups
    int rowTiles = M >> 4;
    int g = blockIdx.x * 8 + wave;
    if (g >= rowTiles * nNG) return;      // uniform per wave: EXEC stays all-1s
    int rt = g / nNG;
    int ng = g - rt * nNG;

    int row = rt * 16 + (lane & 15);      // A-fragment row for this lane
    int kbase = (lane >> 4) * 8;          // 0 or 8
    int nKT = Kpad >> 5;
    const _Float16* ap = A + (size_t)row * Kpad;
    const _Float16* bp = Wp + ((size_t)(ng * 4) * 32 + (size_t)lane) * 16;
    const size_t bstep = (size_t)nNT * 512;   // halves per K-step

    v8f acc0 = {}, acc1 = {}, acc2 = {}, acc3 = {};
    for (int kt = 0; kt < nKT; ++kt) {
        v8h lo = *(const v8h*)(ap + kbase);        // K = kbase..kbase+7
        v8h hi = *(const v8h*)(ap + 16 + kbase);   // K = 16+kbase..16+kbase+7
        v16h a;
#pragma unroll
        for (int i = 0; i < 8; ++i) { a[i] = lo[i]; a[8 + i] = hi[i]; }
        v16h b0 = *(const v16h*)(bp);
        v16h b1 = *(const v16h*)(bp + 512);
        v16h b2 = *(const v16h*)(bp + 1024);
        v16h b3 = *(const v16h*)(bp + 1536);
        acc0 = __builtin_amdgcn_wmma_f32_16x16x32_f16(false, a, false, b0, (short)0, acc0, false, false);
        acc1 = __builtin_amdgcn_wmma_f32_16x16x32_f16(false, a, false, b1, (short)0, acc1, false, false);
        acc2 = __builtin_amdgcn_wmma_f32_16x16x32_f16(false, a, false, b2, (short)0, acc2, false, false);
        acc3 = __builtin_amdgcn_wmma_f32_16x16x32_f16(false, a, false, b3, (short)0, acc3, false, false);
        ap += 32;
        bp += bstep;
    }

    int col = ng * 64 + (lane & 15);
    int rbase = rt * 16 + ((lane >= 16) ? 8 : 0);
    float bv0 = bias ? bias[col]      : 0.0f;
    float bv1 = bias ? bias[col + 16] : 0.0f;
    float bv2 = bias ? bias[col + 32] : 0.0f;
    float bv3 = bias ? bias[col + 48] : 0.0f;
#pragma unroll
    for (int r = 0; r < 8; ++r) {
        float* crow = C + (size_t)(rbase + r) * Nout + col;
        float v0 = acc0[r] + bv0;
        float v1 = acc1[r] + bv1;
        float v2 = acc2[r] + bv2;
        float v3 = acc3[r] + bv3;
        if (leakyFlag) { v0 = leakyf(v0); v1 = leakyf(v1); v2 = leakyf(v2); v3 = leakyf(v3); }
        crow[0]  = v0;
        crow[16] = v1;
        crow[32] = v2;
        crow[48] = v3;
    }
}

// ---------------------------------------------------------------------------
// Edge kernel: msg = leaky(Tnode[src] + edge_attr @ W_edge + mb);
// agg[dst] += msg (float atomics). One thread per (edge, 4 features).
// shiftQ = log2(Nout/4).
// ---------------------------------------------------------------------------
__global__ void edge_msg(const float* __restrict__ T, const int* __restrict__ src,
                         const int* __restrict__ dst, const float* __restrict__ ea,
                         const float* __restrict__ We /*3 x Nout*/,
                         const float* __restrict__ mb, float* __restrict__ agg,
                         int E, int Nout, int shiftQ)
{
    long long idx = (long long)blockIdx.x * blockDim.x + threadIdx.x;
    long long total = (long long)E << shiftQ;
    if (idx >= total) return;
    int j = (int)(idx & (long long)((Nout >> 2) - 1)) * 4;
    long long e = idx >> shiftQ;
    int s = src[e];
    int d = dst[e];
    const float* eap = ea + e * 3;
    float ea0 = eap[0], ea1 = eap[1], ea2 = eap[2];
    float4 t  = *(const float4*)(T + (size_t)s * Nout + j);
    float4 w0 = *(const float4*)(We + j);
    float4 w1 = *(const float4*)(We + Nout + j);
    float4 w2 = *(const float4*)(We + 2 * Nout + j);
    float4 bb = *(const float4*)(mb + j);
    float v0 = leakyf(t.x + ea0 * w0.x + ea1 * w1.x + ea2 * w2.x + bb.x);
    float v1 = leakyf(t.y + ea0 * w0.y + ea1 * w1.y + ea2 * w2.y + bb.y);
    float v2 = leakyf(t.z + ea0 * w0.z + ea1 * w1.z + ea2 * w2.z + bb.z);
    float v3 = leakyf(t.w + ea0 * w0.w + ea1 * w1.w + ea2 * w2.w + bb.w);
    float* ag = agg + (size_t)d * Nout + j;
    atomicAdd(ag + 0, v0);
    atomicAdd(ag + 1, v1);
    atomicAdd(ag + 2, v2);
    atomicAdd(ag + 3, v3);
}

// ---------------------------------------------------------------------------
// Global mean pool (sum part): gsum[batch[n]] += h[n], D = 256
// ---------------------------------------------------------------------------
__global__ void pool_sum(const float* __restrict__ h, const int* __restrict__ batch,
                         float* __restrict__ gsum, int N) {
    long long idx = (long long)blockIdx.x * blockDim.x + threadIdx.x;
    long long total = (long long)N * 64;   // 4 features per thread
    if (idx >= total) return;
    int j = (int)(idx & 63) * 4;
    int node = (int)(idx >> 6);
    float4 hv = *(const float4*)(h + (size_t)node * 256 + j);
    float* gp = gsum + (size_t)batch[node] * 256 + j;
    atomicAdd(gp + 0, hv.x);
    atomicAdd(gp + 1, hv.y);
    atomicAdd(gp + 2, hv.z);
    atomicAdd(gp + 3, hv.w);
}

__global__ void pool_cnt(const int* __restrict__ batch, float* __restrict__ gcnt, int N) {
    int i = blockIdx.x * blockDim.x + threadIdx.x;
    if (i < N) atomicAdd(&gcnt[batch[i]], 1.0f);
}

__global__ void pool_div_f16(const float* __restrict__ gsum, const float* __restrict__ gcnt,
                             _Float16* __restrict__ A, int G) {
    int idx = blockIdx.x * blockDim.x + threadIdx.x;
    if (idx >= G * 256) return;
    int g = idx >> 8;
    A[idx] = (_Float16)(gsum[idx] / fmaxf(gcnt[g], 1.0f));
}

// ---------------------------------------------------------------------------
// Final 64 -> 1 layer (tiny): out[g] = g2[g] . fw2 + fb2
// ---------------------------------------------------------------------------
__global__ void final_out(const float* __restrict__ g2, const float* __restrict__ fw2,
                          const float* __restrict__ fb2, float* __restrict__ out) {
    int g = threadIdx.x;
    if (g < N_GRAPHS) {
        float s = fb2[0];
        for (int j = 0; j < 64; ++j) s += g2[g * 64 + j] * fw2[j];
        out[g] = s;
    }
}

// ---------------------------------------------------------------------------
extern "C" void kernel_launch(void* const* d_in, const int* in_sizes, int n_in,
                              void* d_out, int out_size, void* d_ws, size_t ws_size,
                              hipStream_t stream)
{
    (void)in_sizes; (void)n_in; (void)out_size; (void)ws_size;
    const int N = N_NODES, E = N_EDGES, G = N_GRAPHS;

    const float* x     = (const float*)d_in[0];
    const int*   ei    = (const int*)d_in[1];     // [2, E]: src = ei, dst = ei + E
    const float* ea    = (const float*)d_in[2];
    const int*   batch = (const int*)d_in[3];
    const float* MW[3] = {(const float*)d_in[4],  (const float*)d_in[8],  (const float*)d_in[12]};
    const float* MB[3] = {(const float*)d_in[5],  (const float*)d_in[9],  (const float*)d_in[13]};
    const float* UW[3] = {(const float*)d_in[6],  (const float*)d_in[10], (const float*)d_in[14]};
    const float* UB[3] = {(const float*)d_in[7],  (const float*)d_in[11], (const float*)d_in[15]};
    const float* FW0 = (const float*)d_in[16]; const float* FB0 = (const float*)d_in[17];
    const float* FW1 = (const float*)d_in[18]; const float* FB1 = (const float*)d_in[19];
    const float* FW2 = (const float*)d_in[20]; const float* FB2 = (const float*)d_in[21];

    // ---- workspace carve ----
    char* ws = (char*)d_ws;
    size_t off = 0;
    auto carve = [&](size_t bytes) -> char* {
        char* p = ws + off;
        off += (bytes + 255) & ~(size_t)255;
        return p;
    };
    float*    deg  = (float*)   carve((size_t)N * 4);
    float*    hbuf = (float*)   carve((size_t)N * 256 * 4);
    float*    Tn   = (float*)   carve((size_t)N * 256 * 4);
    float*    agg  = (float*)   carve((size_t)N * 256 * 4);
    _Float16* A16  = (_Float16*)carve((size_t)N * 384 * 2);
    _Float16* Wm   = (_Float16*)carve((size_t)384 * 256 * 2);
    _Float16* Wu   = (_Float16*)carve((size_t)384 * 256 * 2);
    float*    gsum = (float*)   carve((size_t)G * 256 * 4);
    float*    gcnt = (float*)   carve((size_t)G * 4);
    float*    g1   = (float*)   carve((size_t)G * 128 * 4);
    float*    g2   = (float*)   carve((size_t)G * 64 * 4);

    // ---- degrees (computed once; reference reuses same inv_deg every layer) ----
    zero_f32<<<(N + 255) / 256, 256, 0, stream>>>(deg, N);
    deg_count<<<(E + 255) / 256, 256, 0, stream>>>(ei + E, deg, E);

    const int nin[3]  = {10, 64, 128};
    const int nout[3] = {64, 128, 256};
    const int kpm[3]  = {32, 64, 128};    // message GEMM K (node_in padded to 32)
    const int kpu[3]  = {96, 192, 384};   // update GEMM K (out+node_in padded to 32)
    const int shq[3]  = {4, 5, 6};        // log2(nout/4)

    const float* hcur = x;
    for (int l = 0; l < 3; ++l) {
        // 1) pack node-part of message weights (first nin rows of mw)
        {
            int tot = (kpm[l] >> 5) * (nout[l] >> 4) * 32;
            pack_w_f16<<<(tot + 255) / 256, 256, 0, stream>>>(MW[l], nin[l], nout[l], kpm[l], Wm);
        }
        // 2) h -> f16 padded A
        prep_a_f16<<<N, 128, 0, stream>>>(hcur, nin[l], kpm[l], A16);
        // 3) Tnode = h @ mw_node   (no bias / no activation here)
        {
            int gcount = (N >> 4) * (nout[l] >> 6);
            gemm_wmma_f16<<<(gcount + 7) / 8, 256, 0, stream>>>(
                A16, Wm, nullptr, Tn, N, kpm[l], nout[l], 0);
        }
        // 4) scatter messages into agg with float atomics
        zero_f32<<<(int)(((long long)N * nout[l] + 255) / 256), 256, 0, stream>>>(
            agg, (long long)N * nout[l]);
        {
            long long tot = (long long)E << shq[l];
            edge_msg<<<(int)((tot + 255) / 256), 256, 0, stream>>>(
                Tn, ei, ei + E, ea, MW[l] + (size_t)nin[l] * nout[l], MB[l],
                agg, E, nout[l], shq[l]);
        }
        // 5) pack update weights
        {
            int tot = (kpu[l] >> 5) * (nout[l] >> 4) * 32;
            pack_w_f16<<<(tot + 255) / 256, 256, 0, stream>>>(
                UW[l], nout[l] + nin[l], nout[l], kpu[l], Wu);
        }
        // 6) A = [agg*inv_deg | h] -> f16
        prep_au_f16<<<N, 128, 0, stream>>>(agg, deg, hcur, nin[l], nout[l], kpu[l], A16);
        // 7) h_new = leaky(A @ uw + ub)
        {
            int gcount = (N >> 4) * (nout[l] >> 6);
            gemm_wmma_f16<<<(gcount + 7) / 8, 256, 0, stream>>>(
                A16, Wu, UB[l], hbuf, N, kpu[l], nout[l], 1);
        }
        hcur = hbuf;
    }

    // ---- global mean pool ----
    zero_f32<<<(G * 256 + 255) / 256, 256, 0, stream>>>(gsum, G * 256);
    zero_f32<<<1, 64, 0, stream>>>(gcnt, G);
    pool_sum<<<(int)(((long long)N * 64 + 255) / 256), 256, 0, stream>>>(hbuf, batch, gsum, N);
    pool_cnt<<<(N + 255) / 256, 256, 0, stream>>>(batch, gcnt, N);
    pool_div_f16<<<(G * 256 + 255) / 256, 256, 0, stream>>>(gsum, gcnt, A16, G);

    // ---- final MLP: 256 -> 128 -> 64 -> 1 (first two via WMMA, M = 32) ----
    {
        int tot = (256 >> 5) * (128 >> 4) * 32;
        pack_w_f16<<<(tot + 255) / 256, 256, 0, stream>>>(FW0, 256, 128, 256, Wm);
        int gcount = (G >> 4) * (128 >> 6);
        gemm_wmma_f16<<<(gcount + 7) / 8, 256, 0, stream>>>(A16, Wm, FB0, g1, G, 256, 128, 1);
    }
    prep_a_f16<<<G, 128, 0, stream>>>(g1, 128, 128, A16);
    {
        int tot = (128 >> 5) * (64 >> 4) * 32;
        pack_w_f16<<<(tot + 255) / 256, 256, 0, stream>>>(FW1, 128, 64, 128, Wm);
        int gcount = (G >> 4) * (64 >> 6);
        gemm_wmma_f16<<<(gcount + 7) / 8, 256, 0, stream>>>(A16, Wm, FB1, g2, G, 128, 64, 1);
    }
    final_out<<<1, 32, 0, stream>>>(g2, FW2, FB2, (float*)d_out);
}
